// GPT_73358041415931
// MI455X (gfx1250) — matrix operand order repaired and checked by reading, not compiled, not run on hardware
//
#include <hip/hip_runtime.h>
#include <math.h>

// ---------------------------------------------------------------------------
// Types for CDNA5 WMMA (wave32): v_wmma_f32_16x16x32_bf16
// ---------------------------------------------------------------------------
typedef __attribute__((ext_vector_type(16))) __bf16 bf16x16;
typedef __attribute__((ext_vector_type(8)))  float  f32x8;
typedef __attribute__((ext_vector_type(4)))  unsigned int u32x4;

__device__ __forceinline__ f32x8 wmma_bf16(bf16x16 a, bf16x16 b, f32x8 c) {
  // 8 args: (neg_a, A, neg_b, B, c_mod, C, reuse_a, reuse_b)
  return __builtin_amdgcn_wmma_f32_16x16x32_bf16(false, a, false, b, (short)0, c,
                                                 false, false);
}

union FragU { bf16x16 v; u32x4 u[2]; };

// A-matrix 16x32 bf16 fragment. Row M = lane&15 (both lane halves), element e:
//   K = k0 + (lane>>4)*8 + e            for e in [0,8)
//   K = k0 + 16 + (lane>>4)*8 + (e-8)   for e in [8,16)
// -> two contiguous 16B loads at +0 and +16 elements.
__device__ __forceinline__ bf16x16 ldfragA(const __bf16* base, int ld, int k0) {
  int lane = threadIdx.x & 31;
  const __bf16* p = base + (size_t)(lane & 15) * ld + k0 + ((lane >> 4) << 3);
  FragU f;
  f.u[0] = *(const u32x4*)(p);
  f.u[1] = *(const u32x4*)(p + 16);
  return f.v;
}

// B-matrix 32x16 bf16 fragment, memory holds B^T as [N][K] (K contiguous).
// Column N = lane&15, element e -> K = k0 + (lane>>4)*8 + e (e<8), K = k0 + 16 + (lane>>4)*8 + (e-8) (e>=8).
__device__ __forceinline__ bf16x16 ldfragB(const __bf16* baseT, int ld, int k0) {
  int lane = threadIdx.x & 31;
  const __bf16* p = baseT + (size_t)(lane & 15) * ld + k0 + ((lane >> 4) << 3);
  FragU f;
  f.u[0] = *(const u32x4*)(p);
  f.u[1] = *(const u32x4*)(p + 16);
  return f.v;
}

// ---------------------------------------------------------------------------
// Embedding: x[bt, c] = tok_emb[tokens[bt], c] + pos_emb[bt % T, c]
// ---------------------------------------------------------------------------
__global__ __launch_bounds__(256) void embed_kernel(
    const int* __restrict__ tokens, const float* __restrict__ tok_emb,
    const float* __restrict__ pos_emb, float* __restrict__ x) {
  int bt = blockIdx.x;
  int t = bt & 1023;
  int tok = tokens[bt];
  const float* te = tok_emb + (size_t)tok * 1024;
  const float* pe = pos_emb + (size_t)t * 1024;
  float* xr = x + (size_t)bt * 1024;
  for (int c = threadIdx.x; c < 1024; c += 256) xr[c] = te[c] + pe[c];
}

// ---------------------------------------------------------------------------
// LayerNorm -> bf16 output (GEMM A operand)
// ---------------------------------------------------------------------------
__global__ __launch_bounds__(256) void ln_kernel(
    const float* __restrict__ x, const float* __restrict__ w,
    const float* __restrict__ b, __bf16* __restrict__ out, int C) {
  int row = blockIdx.x;
  const float* xr = x + (size_t)row * C;
  __shared__ float ssum[256], ssq[256];
  float s = 0.f, q = 0.f;
  for (int c = threadIdx.x; c < C; c += 256) { float v = xr[c]; s += v; q += v * v; }
  ssum[threadIdx.x] = s; ssq[threadIdx.x] = q;
  __syncthreads();
  for (int o = 128; o > 0; o >>= 1) {
    if ((int)threadIdx.x < o) {
      ssum[threadIdx.x] += ssum[threadIdx.x + o];
      ssq[threadIdx.x]  += ssq[threadIdx.x + o];
    }
    __syncthreads();
  }
  float mu  = ssum[0] / (float)C;
  float var = ssq[0] / (float)C - mu * mu;
  float inv = rsqrtf(var + 1e-5f);
  __bf16* orow = out + (size_t)row * C;
  for (int c = threadIdx.x; c < C; c += 256)
    orow[c] = (__bf16)((xr[c] - mu) * inv * w[c] + b[c]);
}

// ---------------------------------------------------------------------------
// Transpose + fp32->bf16 convert: W[K][N] -> WT[N][K]
// ---------------------------------------------------------------------------
__global__ __launch_bounds__(256) void transpose_cvt_kernel(
    const float* __restrict__ W, __bf16* __restrict__ WT, int K, int N) {
  __shared__ float tile[32][33];
  int k0 = blockIdx.y * 32, n0 = blockIdx.x * 32;
  int tx = threadIdx.x & 31, ty = threadIdx.x >> 5;
  for (int i = ty; i < 32; i += 8)
    tile[i][tx] = W[(size_t)(k0 + i) * N + n0 + tx];
  __syncthreads();
  for (int i = ty; i < 32; i += 8)
    WT[(size_t)(n0 + i) * K + k0 + tx] = (__bf16)tile[tx][i];
}

// head_w [16][101][1024] f32 -> padded bf16 [16][128][1024]
__global__ __launch_bounds__(256) void head_pad_kernel(
    const float* __restrict__ hw, __bf16* __restrict__ out) {
  int eo = blockIdx.x;           // 16*128
  int e = eo >> 7, o = eo & 127;
  __bf16* orow = out + (size_t)eo * 1024;
  if (o < 101) {
    const float* irow = hw + ((size_t)e * 101 + o) * 1024;
    for (int i = threadIdx.x; i < 1024; i += 256) orow[i] = (__bf16)irow[i];
  } else {
    for (int i = threadIdx.x; i < 1024; i += 256) orow[i] = (__bf16)0.0f;
  }
}

// v bf16 [B][T][C] -> vT bf16 [B][H][64][T]
__global__ __launch_bounds__(256) void transpose_v_kernel(
    const __bf16* __restrict__ v, __bf16* __restrict__ vT) {
  __shared__ __bf16 tile[32][34];
  int bh = blockIdx.z;           // b*16 + h
  int b = bh >> 4, h = bh & 15;
  int t0 = blockIdx.x * 32, d0 = blockIdx.y * 32;
  int tx = threadIdx.x & 31, ty = threadIdx.x >> 5;
  for (int i = ty; i < 32; i += 8)
    tile[i][tx] = v[((size_t)(b * 1024 + t0 + i)) * 1024 + h * 64 + d0 + tx];
  __syncthreads();
  for (int i = ty; i < 32; i += 8)
    vT[((size_t)bh * 64 + d0 + i) * 1024 + t0 + tx] = tile[tx][i];
}

// ---------------------------------------------------------------------------
// WMMA GEMM: out = epilogue(A[M,K](bf16) @ BT[N,K]^T(bf16) + bias)
// 4 waves/block; wave tile 64x64 (16 WMMA : 16 b128 loads per k-step);
// block tile 128x128.
// Epilogue: +bias, optional exact GELU, optional +residual, fp32/bf16 stores.
// ---------------------------------------------------------------------------
__global__ __launch_bounds__(128) void gemm_bf16_kernel(
    const __bf16* __restrict__ A, const __bf16* __restrict__ BT,
    const float* __restrict__ bias, const float* __restrict__ res,
    float* __restrict__ outF, __bf16* __restrict__ outB,
    int M, int N, int K, int gelu) {
  int wave = threadIdx.x >> 5;
  int lane = threadIdx.x & 31;
  int wm = wave >> 1, wn = wave & 1;
  int m0 = blockIdx.y * 128 + wm * 64;
  int n0 = blockIdx.x * 128 + wn * 64;

  f32x8 acc[4][4] = {};
  const __bf16* Ab[4];
  const __bf16* Bb[4];
#pragma unroll
  for (int i = 0; i < 4; ++i) {
    Ab[i] = A + (size_t)(m0 + i * 16) * K;
    Bb[i] = BT + (size_t)(n0 + i * 16) * K;
  }

  for (int k0 = 0; k0 < K; k0 += 32) {
    bf16x16 af[4], bfr[4];
#pragma unroll
    for (int i = 0; i < 4; ++i) af[i] = ldfragA(Ab[i], K, k0);
#pragma unroll
    for (int j = 0; j < 4; ++j) bfr[j] = ldfragB(Bb[j], K, k0);
#pragma unroll
    for (int i = 0; i < 4; ++i)
#pragma unroll
      for (int j = 0; j < 4; ++j)
        acc[i][j] = wmma_bf16(af[i], bfr[j], acc[i][j]);
  }

  int col = lane & 15;
  int rb  = (lane >> 4) << 3;
#pragma unroll
  for (int i = 0; i < 4; ++i)
#pragma unroll
    for (int j = 0; j < 4; ++j) {
      f32x8 a = acc[i][j];
      int n = n0 + j * 16 + col;
      float bv = bias[n];
#pragma unroll
      for (int r = 0; r < 8; ++r) {
        int m = m0 + i * 16 + rb + r;
        float v = a[r] + bv;
        if (gelu) v = 0.5f * v * (1.0f + erff(v * 0.70710678118654752f));
        if (res)  v += res[(size_t)m * N + n];
        if (outF) outF[(size_t)m * N + n] = v;
        if (outB) outB[(size_t)m * N + n] = (__bf16)v;
      }
    }
}

// ---------------------------------------------------------------------------
// Attention: one block = (b, h, 16 query rows). 4 waves.
// S = mask(scale * Q K^T) -> LDS (bf16, unnormalized exp after softmax pass)
// O = (P @ V) / rowsum via WMMA with pre-transposed V^T.
// ---------------------------------------------------------------------------
__global__ __launch_bounds__(128) void attn_kernel(
    const __bf16* __restrict__ Q, const __bf16* __restrict__ Km,
    const __bf16* __restrict__ Vt, __bf16* __restrict__ Y) {
  const int PITCH = 1040;                         // bf16 elems; 2080B row, 16B aligned
  __shared__ __align__(16) __bf16 sc[16 * PITCH];
  __shared__ float red[16][8];
  __shared__ float rowmax[16], rowsum[16];

  int q0 = blockIdx.x * 16;
  int h  = blockIdx.y;
  int b  = blockIdx.z;
  int wave = threadIdx.x >> 5;
  int lane = threadIdx.x & 31;
  const float scale = 0.125f;                     // 1/sqrt(64)

  const __bf16* qbase = Q + ((size_t)(b * 1024 + q0)) * 1024 + h * 64;
  const __bf16* kbase = Km + ((size_t)b * 1024) * 1024 + h * 64;

  // --- scores: wave covers keys [wave*256, wave*256+256) ---
  for (int kt = wave * 256; kt < wave * 256 + 256; kt += 16) {
    f32x8 acc = {};
    bf16x16 af = ldfragA(qbase, 1024, 0);
    bf16x16 bf0 = ldfragB(kbase + (size_t)kt * 1024, 1024, 0);
    acc = wmma_bf16(af, bf0, acc);
    af  = ldfragA(qbase, 1024, 32);
    bf0 = ldfragB(kbase + (size_t)kt * 1024, 1024, 32);
    acc = wmma_bf16(af, bf0, acc);

    int col = lane & 15;
    int rb  = (lane >> 4) << 3;
    int tk  = kt + col;
    bool colok = ((tk & 15) != 15);               // (t_k % 16) != 15
#pragma unroll
    for (int r = 0; r < 8; ++r) {
      int tq = q0 + rb + r;
      float v = (colok && (tk <= tq)) ? acc[r] * scale : -1e30f;
      sc[(rb + r) * PITCH + tk] = (__bf16)v;
    }
  }
  __syncthreads();

  // --- softmax (unnormalized exp in LDS + rowsum) ---
  int rr = threadIdx.x >> 3;                      // row 0..15
  int ch = threadIdx.x & 7;                       // 8 chunks of 128
  __bf16* srow = sc + rr * PITCH + ch * 128;
  float m = -1e30f;
  for (int i = 0; i < 128; ++i) m = fmaxf(m, (float)srow[i]);
  red[rr][ch] = m;
  __syncthreads();
  if (ch == 0) {
    float mm = red[rr][0];
    for (int i = 1; i < 8; ++i) mm = fmaxf(mm, red[rr][i]);
    rowmax[rr] = mm;
  }
  __syncthreads();
  float rmx = rowmax[rr];
  float ssum = 0.f;
  for (int i = 0; i < 128; ++i) {
    float e = __expf((float)srow[i] - rmx);
    srow[i] = (__bf16)e;
    ssum += e;
  }
  red[rr][ch] = ssum;
  __syncthreads();
  if (ch == 0) {
    float t = 0.f;
    for (int i = 0; i < 8; ++i) t += red[rr][i];
    rowsum[rr] = t;
  }
  __syncthreads();

  // --- O = P @ V, wave owns d block [wave*16, wave*16+16) ---
  f32x8 acc = {};
  const __bf16* vbase = Vt + (((size_t)(b * 16 + h)) * 64 + wave * 16) * 1024;
  for (int kk = 0; kk < 1024; kk += 32) {
    bf16x16 af  = ldfragA(sc, PITCH, kk);
    bf16x16 bf0 = ldfragB(vbase, 1024, kk);
    acc = wmma_bf16(af, bf0, acc);
  }
  int col = lane & 15;
  int rb  = (lane >> 4) << 3;
#pragma unroll
  for (int r = 0; r < 8; ++r) {
    int row = rb + r;
    float inv = 1.0f / rowsum[row];
    Y[((size_t)(b * 1024 + q0 + row)) * 1024 + h * 64 + wave * 16 + col] =
        (__bf16)(acc[r] * inv);
  }
}

// ---------------------------------------------------------------------------
// EinLinear head: logits[s*16+e, o] = sum_i headT[e][o][i] * x[s*16+e, i]
// A rows strided by 16*C (the [*,16,C] reshape). N padded to 128, store o<101.
// ---------------------------------------------------------------------------
__global__ __launch_bounds__(256) void head_kernel(
    const __bf16* __restrict__ xb, const __bf16* __restrict__ headT,
    float* __restrict__ out) {
  int e = blockIdx.z;
  int wave = threadIdx.x >> 5;
  int lane = threadIdx.x & 31;
  int wm = wave >> 1, wn = wave & 1;
  int m0 = wm * 32;                               // M=128 covered by 4 waves
  int n0 = blockIdx.x * 64 + wn * 32;             // N=128 via gridDim.x=2
  const int LDA = 16 * 1024;

  const __bf16* A0 = xb + (size_t)e * 1024 + (size_t)m0 * LDA;
  const __bf16* A1 = xb + (size_t)e * 1024 + (size_t)(m0 + 16) * LDA;
  const __bf16* B0 = headT + (size_t)e * 128 * 1024 + (size_t)n0 * 1024;
  const __bf16* B1 = headT + (size_t)e * 128 * 1024 + (size_t)(n0 + 16) * 1024;

  f32x8 acc00 = {}, acc01 = {}, acc10 = {}, acc11 = {};
  for (int k0 = 0; k0 < 1024; k0 += 32) {
    bf16x16 a0 = ldfragA(A0, LDA, k0);
    bf16x16 a1 = ldfragA(A1, LDA, k0);
    bf16x16 b0 = ldfragB(B0, 1024, k0);
    bf16x16 b1 = ldfragB(B1, 1024, k0);
    acc00 = wmma_bf16(a0, b0, acc00);
    acc01 = wmma_bf16(a0, b1, acc01);
    acc10 = wmma_bf16(a1, b0, acc10);
    acc11 = wmma_bf16(a1, b1, acc11);
  }
  int col = lane & 15;
  int rb  = (lane >> 4) << 3;
  f32x8 accs[4] = {acc00, acc01, acc10, acc11};
#pragma unroll
  for (int im = 0; im < 2; ++im)
#pragma unroll
    for (int jn = 0; jn < 2; ++jn) {
      int o = n0 + jn * 16 + col;
      if (o < 101) {
        f32x8 a = accs[im * 2 + jn];
#pragma unroll
        for (int r = 0; r < 8; ++r) {
          int s = m0 + im * 16 + rb + r;
          out[((size_t)(s * 16 + e)) * 101 + o] = a[r];
        }
      }
    }
}

// ---------------------------------------------------------------------------
// Host driver
// ---------------------------------------------------------------------------
extern "C" void kernel_launch(void* const* d_in, const int* in_sizes, int n_in,
                              void* d_out, int out_size, void* d_ws, size_t ws_size,
                              hipStream_t stream) {
  (void)in_sizes; (void)n_in; (void)out_size; (void)ws_size;
  const int*   tokens  = (const int*)  d_in[0];
  const float* tok_emb = (const float*)d_in[1];
  const float* pos_emb = (const float*)d_in[2];
  const float* Wq = (const float*)d_in[3];  const float* bq = (const float*)d_in[4];
  const float* Wk = (const float*)d_in[5];  const float* bk = (const float*)d_in[6];
  const float* Wv = (const float*)d_in[7];  const float* bv = (const float*)d_in[8];
  const float* Wp = (const float*)d_in[9];  const float* bp = (const float*)d_in[10];
  const float* ln1w = (const float*)d_in[11]; const float* ln1b = (const float*)d_in[12];
  const float* ln2w = (const float*)d_in[13]; const float* ln2b = (const float*)d_in[14];
  const float* W1 = (const float*)d_in[15]; const float* b1 = (const float*)d_in[16];
  const float* W2 = (const float*)d_in[17]; const float* b2 = (const float*)d_in[18];
  const float* lnfw = (const float*)d_in[19]; const float* lnfb = (const float*)d_in[20];
  const float* head_w = (const float*)d_in[21];
  float* out = (float*)d_out;

  const int B = 2, T = 1024, C = 1024, L = 4, F = 4096;
  const int M = B * T;  // 2048

  // ---- carve workspace (all sizes 256B multiples) ----
  char* p = (char*)d_ws;
  auto carve = [&](size_t bytes) -> char* { char* r = p; p += bytes; return r; };
  float*  x    = (float*) carve((size_t)M * C * 4);   // residual stream fp32
  __bf16* hb   = (__bf16*)carve((size_t)M * C * 2);   // LN output bf16
  __bf16* qb   = (__bf16*)carve((size_t)M * C * 2);
  __bf16* kb   = (__bf16*)carve((size_t)M * C * 2);
  __bf16* vb   = (__bf16*)carve((size_t)M * C * 2);
  __bf16* vT   = (__bf16*)carve((size_t)M * C * 2);   // [B][H][64][T]
  __bf16* ybuf = (__bf16*)carve((size_t)M * C * 2);   // attn out
  __bf16* mid  = (__bf16*)carve((size_t)M * F * 2);   // MLP hidden
  __bf16* WqT  = (__bf16*)carve((size_t)L * C * C * 2);
  __bf16* WkT  = (__bf16*)carve((size_t)L * C * C * 2);
  __bf16* WvT  = (__bf16*)carve((size_t)L * C * C * 2);
  __bf16* WpT  = (__bf16*)carve((size_t)L * C * C * 2);
  __bf16* W1T  = (__bf16*)carve((size_t)L * C * F * 2);
  __bf16* W2T  = (__bf16*)carve((size_t)L * C * F * 2);
  __bf16* hT   = (__bf16*)carve((size_t)16 * 128 * C * 2);

  // ---- weight prep: transpose + convert to bf16 [N][K] ----
  for (int i = 0; i < L; ++i) {
    size_t wo = (size_t)i * C * C;
    size_t fo = (size_t)i * C * F;
    transpose_cvt_kernel<<<dim3(32, 32), 256, 0, stream>>>(Wq + wo, WqT + wo, C, C);
    transpose_cvt_kernel<<<dim3(32, 32), 256, 0, stream>>>(Wk + wo, WkT + wo, C, C);
    transpose_cvt_kernel<<<dim3(32, 32), 256, 0, stream>>>(Wv + wo, WvT + wo, C, C);
    transpose_cvt_kernel<<<dim3(32, 32), 256, 0, stream>>>(Wp + wo, WpT + wo, C, C);
    transpose_cvt_kernel<<<dim3(128, 32), 256, 0, stream>>>(W1 + fo, W1T + fo, C, F);
    transpose_cvt_kernel<<<dim3(32, 128), 256, 0, stream>>>(W2 + fo, W2T + fo, F, C);
  }
  head_pad_kernel<<<16 * 128, 256, 0, stream>>>(head_w, hT);

  // ---- embedding ----
  embed_kernel<<<M, 256, 0, stream>>>(tokens, tok_emb, pos_emb, x);

  // ---- transformer layers ----
  for (int i = 0; i < L; ++i) {
    size_t wo = (size_t)i * C * C;
    size_t fo = (size_t)i * C * F;
    ln_kernel<<<M, 256, 0, stream>>>(x, ln1w + i * C, ln1b + i * C, hb, C);
    gemm_bf16_kernel<<<dim3(C / 128, M / 128), 128, 0, stream>>>(
        hb, WqT + wo, bq + i * C, nullptr, nullptr, qb, M, C, C, 0);
    gemm_bf16_kernel<<<dim3(C / 128, M / 128), 128, 0, stream>>>(
        hb, WkT + wo, bk + i * C, nullptr, nullptr, kb, M, C, C, 0);
    gemm_bf16_kernel<<<dim3(C / 128, M / 128), 128, 0, stream>>>(
        hb, WvT + wo, bv + i * C, nullptr, nullptr, vb, M, C, C, 0);
    transpose_v_kernel<<<dim3(32, 2, 32), 256, 0, stream>>>(vb, vT);
    attn_kernel<<<dim3(64, 16, 2), 128, 0, stream>>>(qb, kb, vT, ybuf);
    gemm_bf16_kernel<<<dim3(C / 128, M / 128), 128, 0, stream>>>(
        ybuf, WpT + wo, bp + i * C, x, x, nullptr, M, C, C, 0);
    ln_kernel<<<M, 256, 0, stream>>>(x, ln2w + i * C, ln2b + i * C, hb, C);
    gemm_bf16_kernel<<<dim3(F / 128, M / 128), 128, 0, stream>>>(
        hb, W1T + fo, b1 + i * F, nullptr, nullptr, mid, M, F, C, 1);
    gemm_bf16_kernel<<<dim3(C / 128, M / 128), 128, 0, stream>>>(
        mid, W2T + fo, b2 + i * C, x, x, nullptr, M, C, F, 0);
  }

  // ---- final LN + EinLinear head ----
  ln_kernel<<<M, 256, 0, stream>>>(x, lnfw, lnfb, hb, C);
  head_kernel<<<dim3(2, 1, 16), 256, 0, stream>>>(hb, hT, out);
}
